// GIN_90503550861610
// MI455X (gfx1250) — matrix operand order repaired and checked by reading, not compiled, not run on hardware
//
#include <hip/hip_runtime.h>

#define N_NODES 100000
#define N_EDGES 1600000
#define G_GRAPHS 512
#define FDIM 128
#define TDIM 10
#define BN_EPS 1e-5f

typedef __bf16 bf16_t;
typedef __attribute__((ext_vector_type(4)))  __bf16 v4bf;
typedef __attribute__((ext_vector_type(8)))  __bf16 v8bf;
typedef __attribute__((ext_vector_type(16))) __bf16 v16bf;
typedef __attribute__((ext_vector_type(8)))  float  v8f;

// ---------------- utility ----------------
__global__ void k_zero(float* __restrict__ p, long n) {
    long i = (long)blockIdx.x * blockDim.x + threadIdx.x;
    long stride = (long)gridDim.x * blockDim.x;
    for (; i < n; i += stride) p[i] = 0.f;
}

// Convert + transpose weights: Wbt[m][n][k] = bf16(W[m/2][k][n]); m = l*2 + which
__global__ void k_cvt_weights(const float* __restrict__ Ws1, const float* __restrict__ Ws2,
                              bf16_t* __restrict__ Wbt) {
    int idx = blockIdx.x * blockDim.x + threadIdx.x;
    if (idx >= 6 * FDIM * FDIM) return;
    int m   = idx >> 14;
    int rem = idx & 16383;
    int n   = rem >> 7;
    int k   = rem & 127;
    int l   = m >> 1;
    const float* src = (m & 1) ? Ws2 : Ws1;
    float v = src[((size_t)l * FDIM + k) * FDIM + n];
    Wbt[((size_t)m << 14) + (n << 7) + k] = (bf16_t)v;
}

// vectorized: n4 = count of float4 groups
__global__ void k_cvt_x(const float* __restrict__ x, float* __restrict__ hf,
                        bf16_t* __restrict__ hb, long n4) {
    long i = (long)blockIdx.x * blockDim.x + threadIdx.x;
    long stride = (long)gridDim.x * blockDim.x;
    for (; i < n4; i += stride) {
        float4 v = ((const float4*)x)[i];
        ((float4*)hf)[i] = v;
        v4bf o; o[0] = (bf16_t)v.x; o[1] = (bf16_t)v.y; o[2] = (bf16_t)v.z; o[3] = (bf16_t)v.w;
        ((v4bf*)hb)[i] = o;
    }
}

// ---------------- WMMA GEMM: out[N,128] = A[N,128](bf16) @ W + bias ----------------
// Wt is [n][k] bf16 (transposed, n-major). Block: 256 thr = 8 waves, 256 rows.
// Each wave: 2 M-tiles x 8 N-tiles -> every B fragment feeds 2 independent WMMAs.
__global__ __launch_bounds__(256)
void k_gemm_bias(const bf16_t* __restrict__ A, const bf16_t* __restrict__ Wt,
                 const float* __restrict__ bias, float* __restrict__ out, int nrows) {
    __shared__ uint4 Wl4[2048];                 // 32 KB: full 128x128 bf16 weight tile
    bf16_t* Wl = (bf16_t*)Wl4;
    const uint4* wsrc = (const uint4*)Wt;
    for (int i = threadIdx.x; i < 2048; i += 256) Wl4[i] = wsrc[i];
    __syncthreads();

    int wave = threadIdx.x >> 5;
    int lane = threadIdx.x & 31;
    int hi   = lane >> 4;                       // half-wave selector
    int l15  = lane & 15;
    int m0   = blockIdx.x * 256 + wave * 32;

    // Two A row-tiles per wave; both lane halves cover M=0..15 (hi half holds K+8 block)
    int r0 = m0 + l15;        if (r0 >= nrows) r0 = nrows - 1;
    int r1 = m0 + 16 + l15;   if (r1 >= nrows) r1 = nrows - 1;
    const bf16_t* a0p = A + (size_t)r0 * FDIM + hi * 8;
    const bf16_t* a1p = A + (size_t)r1 * FDIM + hi * 8;

    v8f acc[2][8];
    #pragma unroll
    for (int mt = 0; mt < 2; mt++)
        #pragma unroll
        for (int t = 0; t < 8; t++)
            #pragma unroll
            for (int r = 0; r < 8; r++) acc[mt][t][r] = 0.f;

    #pragma unroll
    for (int kk = 0; kk < 4; kk++) {            // K = 128 in steps of 32
        int k0 = kk * 32;
        v8bf a0lo = *(const v8bf*)(a0p + k0);
        v8bf a0hi = *(const v8bf*)(a0p + k0 + 16);
        v16bf a0 = __builtin_shufflevector(a0lo, a0hi,
                     0,1,2,3,4,5,6,7,8,9,10,11,12,13,14,15);
        v8bf a1lo = *(const v8bf*)(a1p + k0);
        v8bf a1hi = *(const v8bf*)(a1p + k0 + 16);
        v16bf a1 = __builtin_shufflevector(a1lo, a1hi,
                     0,1,2,3,4,5,6,7,8,9,10,11,12,13,14,15);
        #pragma unroll
        for (int nt = 0; nt < 8; nt++) {        // 8 N-tiles of 16
            const bf16_t* bp = Wl + ((nt * 16 + l15) << 7) + k0 + hi * 8;
            v8bf blo = *(const v8bf*)(bp);
            v8bf bhi = *(const v8bf*)(bp + 16);
            v16bf b = __builtin_shufflevector(blo, bhi,
                     0,1,2,3,4,5,6,7,8,9,10,11,12,13,14,15);
            acc[0][nt] = __builtin_amdgcn_wmma_f32_16x16x32_bf16(
                false, a0, false, b, (short)0, acc[0][nt], false, false);
            acc[1][nt] = __builtin_amdgcn_wmma_f32_16x16x32_bf16(
                false, a1, false, b, (short)0, acc[1][nt], false, false);
        }
    }

    #pragma unroll
    for (int mt = 0; mt < 2; mt++) {
        #pragma unroll
        for (int nt = 0; nt < 8; nt++) {
            int n = nt * 16 + l15;
            float bv = bias[n];
            #pragma unroll
            for (int r = 0; r < 8; r++) {
                int m = m0 + mt * 16 + hi * 8 + r;   // C layout: lo lanes M=r, hi lanes M=8+r
                if (m < nrows) out[(size_t)m * FDIM + n] = acc[mt][nt][r] + bv;
            }
        }
    }
}

// ---------------- BatchNorm ----------------
__global__ void k_bn_stats(const float* __restrict__ tmp, float* __restrict__ cs, int nrows) {
    __shared__ float ls[256];
    ls[threadIdx.x] = 0.f;
    __syncthreads();
    int c  = threadIdx.x & 127;
    int r0 = blockIdx.x * 64 + (threadIdx.x >> 7);
    int re = blockIdx.x * 64 + 64;
    if (re > nrows) re = nrows;
    float s = 0.f, q = 0.f;
    for (int r = r0; r < re; r += 2) {
        float v = tmp[(size_t)r * 128 + c];
        s += v; q += v * v;
    }
    atomicAdd(&ls[c], s);
    atomicAdd(&ls[128 + c], q);
    __syncthreads();
    atomicAdd(&cs[threadIdx.x], ls[threadIdx.x]);
}

__global__ void k_bn_finalize(const float* __restrict__ cs, const float* __restrict__ gamma,
                              const float* __restrict__ beta, float* __restrict__ scsh) {
    int c = threadIdx.x;                        // 128 threads
    float inv_n = 1.f / (float)N_NODES;
    float mean = cs[c] * inv_n;
    float var  = cs[128 + c] * inv_n - mean * mean;
    float sc   = gamma[c] * rsqrtf(var + BN_EPS);
    scsh[c]       = sc;
    scsh[128 + c] = beta[c] - mean * sc;
}

// vectorized over float4; column index derived from (i*4) & 127
__global__ void k_bn_apply(const float* __restrict__ tmp, const float* __restrict__ scsh,
                           float* __restrict__ hf, bf16_t* __restrict__ hb, long n4) {
    long i = (long)blockIdx.x * blockDim.x + threadIdx.x;
    long stride = (long)gridDim.x * blockDim.x;
    for (; i < n4; i += stride) {
        int c = (int)((i << 2) & 127);
        float4 v = ((const float4*)tmp)[i];
        float4 o;
        o.x = fmaxf(v.x * scsh[c + 0] + scsh[128 + c + 0], 0.f);
        o.y = fmaxf(v.y * scsh[c + 1] + scsh[128 + c + 1], 0.f);
        o.z = fmaxf(v.z * scsh[c + 2] + scsh[128 + c + 2], 0.f);
        o.w = fmaxf(v.w * scsh[c + 3] + scsh[128 + c + 3], 0.f);
        ((float4*)hf)[i] = o;
        v4bf ob; ob[0] = (bf16_t)o.x; ob[1] = (bf16_t)o.y; ob[2] = (bf16_t)o.z; ob[3] = (bf16_t)o.w;
        ((v4bf*)hb)[i] = ob;
    }
}

// ---------------- GIN aggregation ----------------
__global__ void k_scatter(const int* __restrict__ ei, const float* __restrict__ hf,
                          float* __restrict__ agg) {
    long gid = (long)blockIdx.x * 256 + threadIdx.x;
    if (gid >= (long)N_EDGES * 32) return;
    int e = (int)(gid >> 5);
    int c = (int)((gid & 31) << 2);
    int s = ei[e];
    int d = ei[N_EDGES + e];
    const float4 v = *(const float4*)(hf + (size_t)s * 128 + c);
    float* ap = agg + (size_t)d * 128 + c;
    atomicAdd(ap + 0, v.x); atomicAdd(ap + 1, v.y);
    atomicAdd(ap + 2, v.z); atomicAdd(ap + 3, v.w);
}

__global__ void k_add_agg(const float* __restrict__ agg, float* __restrict__ hf,
                          bf16_t* __restrict__ hb, long n4) {
    long i = (long)blockIdx.x * blockDim.x + threadIdx.x;
    long stride = (long)gridDim.x * blockDim.x;
    for (; i < n4; i += stride) {
        float4 a = ((const float4*)agg)[i];
        float4 h = ((const float4*)hf)[i];
        float4 o; o.x = h.x + a.x; o.y = h.y + a.y; o.z = h.z + a.z; o.w = h.w + a.w;
        ((float4*)hf)[i] = o;
        v4bf ob; ob[0] = (bf16_t)o.x; ob[1] = (bf16_t)o.y; ob[2] = (bf16_t)o.z; ob[3] = (bf16_t)o.w;
        ((v4bf*)hb)[i] = ob;
    }
}

// ---------------- pooling + readout ----------------
__global__ void k_count(const int* __restrict__ batch, float* __restrict__ cnt) {
    int i = blockIdx.x * 256 + threadIdx.x;
    if (i < N_NODES) atomicAdd(&cnt[batch[i]], 1.f);
}

__global__ void k_pool(const float* __restrict__ hf, const int* __restrict__ batch,
                       float* __restrict__ psum) {
    long gid = (long)blockIdx.x * 256 + threadIdx.x;
    if (gid >= (long)N_NODES * 32) return;
    int r = (int)(gid >> 5);
    int c = (int)((gid & 31) << 2);
    int g = batch[r];
    const float4 v = *(const float4*)(hf + (size_t)r * 128 + c);
    float* pp = psum + (size_t)g * 128 + c;
    atomicAdd(pp + 0, v.x); atomicAdd(pp + 1, v.y);
    atomicAdd(pp + 2, v.z); atomicAdd(pp + 3, v.w);
}

// out[g,t] += (psum[g]/max(cnt,1)) . linW[:,t] + linb[t]   (mean-pool commutes w/ linear)
__global__ void k_readout(const float* __restrict__ psum, const float* __restrict__ cnt,
                          const float* __restrict__ linW, const float* __restrict__ linb,
                          float* __restrict__ out) {
    int gid = blockIdx.x * blockDim.x + threadIdx.x;
    if (gid >= G_GRAPHS * TDIM) return;
    int g = gid / TDIM, t = gid % TDIM;
    float s = 0.f;
    #pragma unroll 4
    for (int k = 0; k < 128; k++) s += psum[(size_t)g * 128 + k] * linW[k * TDIM + t];
    float c = cnt[g]; if (c < 1.f) c = 1.f;
    out[gid] += s / c + linb[t];
}

// ---------------- launch ----------------
extern "C" void kernel_launch(void* const* d_in, const int* in_sizes, int n_in,
                              void* d_out, int out_size, void* d_ws, size_t ws_size,
                              hipStream_t stream) {
    const float* x    = (const float*)d_in[0];
    const int*   ei   = (const int*)  d_in[1];
    const int*   batch= (const int*)  d_in[2];
    const float* Ws1  = (const float*)d_in[3];
    const float* bs1  = (const float*)d_in[4];
    const float* g1   = (const float*)d_in[5];
    const float* be1  = (const float*)d_in[6];
    const float* Ws2  = (const float*)d_in[7];
    const float* bs2  = (const float*)d_in[8];
    const float* g2   = (const float*)d_in[9];
    const float* be2  = (const float*)d_in[10];
    const float* linW = (const float*)d_in[11];
    const float* linb = (const float*)d_in[12];
    float* out = (float*)d_out;

    char* p = (char*)d_ws;
    auto take = [&](size_t bytes) { char* r = p; p += (bytes + 255) & ~(size_t)255; return r; };
    bf16_t* hb   = (bf16_t*)take((size_t)N_NODES * 128 * 2);
    float*  hf   = (float*) take((size_t)N_NODES * 128 * 4);
    float*  tmp  = (float*) take((size_t)N_NODES * 128 * 4);
    float*  agg  = tmp;                               // disjoint lifetimes -> alias
    bf16_t* Wbt  = (bf16_t*)take((size_t)6 * 16384 * 2);
    float*  cs   = (float*) take(256 * 4);
    float*  scsh = (float*) take(256 * 4);
    float*  psum = (float*) take((size_t)G_GRAPHS * 128 * 4);
    float*  cnt  = (float*) take(G_GRAPHS * 4);

    const long NH  = (long)N_NODES * 128;
    const long NH4 = NH / 4;
    const int gemm_blocks = (N_NODES + 255) / 256;            // 391
    const int ew_blocks   = 2048;                              // grid-stride elementwise
    const int sc_blocks   = (int)(((long)N_EDGES * 32 + 255) / 256);
    const int pl_blocks   = (int)(((long)N_NODES * 32 + 255) / 256);

    k_zero<<<32, 256, 0, stream>>>(out, (long)G_GRAPHS * TDIM);
    k_zero<<<4, 256, 0, stream>>>(cnt, G_GRAPHS);
    k_count<<<(N_NODES + 255) / 256, 256, 0, stream>>>(batch, cnt);
    k_cvt_weights<<<(6 * 16384 + 255) / 256, 256, 0, stream>>>(Ws1, Ws2, Wbt);
    k_cvt_x<<<ew_blocks, 256, 0, stream>>>(x, hf, hb, NH4);

    for (int l = 0; l < 3; l++) {
        if (l > 0) {
            k_zero<<<4096, 256, 0, stream>>>(agg, NH);
            k_scatter<<<sc_blocks, 256, 0, stream>>>(ei, hf, agg);
            k_add_agg<<<ew_blocks, 256, 0, stream>>>(agg, hf, hb, NH4);
        }
        // Linear1 -> BN -> ReLU
        k_gemm_bias<<<gemm_blocks, 256, 0, stream>>>(hb, Wbt + (size_t)(l * 2 + 0) * 16384,
                                                     bs1 + l * 128, tmp, N_NODES);
        k_zero<<<1, 256, 0, stream>>>(cs, 256);
        k_bn_stats<<<(N_NODES + 63) / 64, 256, 0, stream>>>(tmp, cs, N_NODES);
        k_bn_finalize<<<1, 128, 0, stream>>>(cs, g1 + l * 128, be1 + l * 128, scsh);
        k_bn_apply<<<ew_blocks, 256, 0, stream>>>(tmp, scsh, hf, hb, NH4);
        // Linear2 -> BN -> ReLU
        k_gemm_bias<<<gemm_blocks, 256, 0, stream>>>(hb, Wbt + (size_t)(l * 2 + 1) * 16384,
                                                     bs2 + l * 128, tmp, N_NODES);
        k_zero<<<1, 256, 0, stream>>>(cs, 256);
        k_bn_stats<<<(N_NODES + 63) / 64, 256, 0, stream>>>(tmp, cs, N_NODES);
        k_bn_finalize<<<1, 128, 0, stream>>>(cs, g2 + l * 128, be2 + l * 128, scsh);
        k_bn_apply<<<ew_blocks, 256, 0, stream>>>(tmp, scsh, hf, hb, NH4);
        // Readout: out += mean_pool(h) @ linW[l] + linb[l]
        k_zero<<<256, 256, 0, stream>>>(psum, (long)G_GRAPHS * 128);
        k_pool<<<pl_blocks, 256, 0, stream>>>(hf, batch, psum);
        k_readout<<<(G_GRAPHS * TDIM + 255) / 256, 256, 0, stream>>>(
            psum, cnt, linW + (size_t)l * 128 * TDIM, linb + l * TDIM, out);
    }
}
